// Transformer_89361089560851
// MI455X (gfx1250) — matrix operand order repaired and checked by reading, not compiled, not run on hardware
//
#include <hip/hip_runtime.h>
#include <math.h>

// ---------------------------------------------------------------------------
// Transformer (encoder-decoder) for MI455X / gfx1250.
// All GEMMs via V_WMMA_F32_16X16X32_BF16 (fp32 data converted to bf16 tiles in
// LDS, fp32 accumulation). wave32, 256-thread blocks = 8 waves.
// Block tile 128x64; each wave computes a 16x64 strip (4 WMMAs / K-step,
// single shared A fragment). Register-staged software pipeline: globals for
// K-step i+1 are loaded AND converted to packed bf16 registers while WMMAs
// for step i execute (12 staged VGPRs -> no spills). TRANSB/GUARD template
// params keep the hot loop branch-free. All call sites have M % 128 == 0 and
// K % 32 == 0; GUARD=1 only for the V=10000 generator GEMM.
// ---------------------------------------------------------------------------

#define D_MODEL 512
#define N_HEADS 8
#define DK_HEAD 64
#define FF_DIM  2048
#define VOCAB   10000
#define NLAYER  6
#define ATT_DIM 2048
#define BATCH   4
#define SEQ     512
#define LN_EPS  1e-6f

typedef __attribute__((ext_vector_type(16))) __bf16 v16bf;
typedef __attribute__((ext_vector_type(8)))  __bf16 v8bf;
typedef __attribute__((ext_vector_type(8)))  float  v8f;

__device__ __forceinline__ __bf16 f2bf(float f) {
  unsigned u = __builtin_bit_cast(unsigned, f);
  unsigned short s = (unsigned short)(u >> 16);   // truncate to bf16
  return __builtin_bit_cast(__bf16, s);
}

__device__ __forceinline__ v8bf pack8(float4 a, float4 b) {
  v8bf p;
  p[0] = f2bf(a.x); p[1] = f2bf(a.y); p[2] = f2bf(a.z); p[3] = f2bf(a.w);
  p[4] = f2bf(b.x); p[5] = f2bf(b.y); p[6] = f2bf(b.z); p[7] = f2bf(b.w);
  return p;
}

__device__ __forceinline__ v16bf cat8(v8bf a, v8bf b) {
  return __builtin_shufflevector(a, b, 0,1,2,3,4,5,6,7,8,9,10,11,12,13,14,15);
}

#define TILE_M 128
#define TILE_N 64
#define TILE_K 32
#define LDS_PAD 16   // row stride 48 bf16 = 96B (16B aligned, spreads banks)

template<int TRANSB, int GUARD>
__global__ __launch_bounds__(256)
void gemm_wmma_kernel(const float* __restrict__ A, int lda, long long sAb, long long sAh,
                      const float* __restrict__ B, int ldb, long long sBb, long long sBh,
                      float* __restrict__ C, int ldc, long long sCb, long long sCh,
                      const float* __restrict__ bias,
                      const float* __restrict__ Res,
                      int M, int N, int K, int batchH,
                      float scale, int relu)
{
  __shared__ __bf16 As[TILE_M][TILE_K + LDS_PAD];
  __shared__ __bf16 Bs[TILE_N][TILE_K + LDS_PAD];   // stored [n][k] always

  const int z  = blockIdx.z;
  const int zb = z / batchH, zh = z % batchH;
  const float* Ap = A + (long long)zb * sAb + (long long)zh * sAh;
  const float* Bp = B + (long long)zb * sBb + (long long)zh * sBh;
  float*       Cp = C + (long long)zb * sCb + (long long)zh * sCh;

  const int blockM = blockIdx.y * TILE_M;
  const int blockN = blockIdx.x * TILE_N;

  const int t    = threadIdx.x;
  const int wave = t >> 5;         // 0..7 -> M strip wave*16
  const int lane = t & 31;
  const int mrow = wave * 16;

  v8f acc0 = {}, acc1 = {}, acc2 = {}, acc3 = {};

  // staging indices
  const int arow = t >> 1;          // 0..127
  const int acb  = (t & 1) * 16;    // 0,16
  const int brow = t >> 2;          // 0..63
  const int bcb  = (t & 3) * 8;     // 0,8,16,24
  const int gn_stage = blockN + brow;
  const bool bok = !GUARD || (gn_stage < N);

  // register staging (one K-step ahead, already bf16-packed: 12 VGPRs total)
  v8bf pa0, pa1, pb;

  auto load_tile = [&](int k0) {
    const float4* a4 = (const float4*)(Ap + (long long)(blockM + arow) * lda + k0 + acb);
    float4 a0 = a4[0], a1 = a4[1], a2 = a4[2], a3 = a4[3];
    pa0 = pack8(a0, a1);
    pa1 = pack8(a2, a3);
    if (TRANSB) {
      float4 b0 = make_float4(0.f, 0.f, 0.f, 0.f), b1 = b0;
      if (bok) {
        const float4* b4 = (const float4*)(Bp + (long long)gn_stage * ldb + k0 + bcb);
        b0 = b4[0]; b1 = b4[1];
      }
      pb = pack8(b0, b1);
    } else {
      v8bf p;
      #pragma unroll
      for (int j = 0; j < 8; ++j)
        p[j] = f2bf(Bp[(long long)(k0 + bcb + j) * ldb + gn_stage]);
      pb = p;
    }
  };

  load_tile(0);

  for (int k0 = 0; k0 < K; k0 += TILE_K) {
    // ---- commit staged bf16 registers to LDS (pure ds_store_b128) ----
    *(v8bf*)&As[arow][acb]     = pa0;
    *(v8bf*)&As[arow][acb + 8] = pa1;
    *(v8bf*)&Bs[brow][bcb]     = pb;
    __syncthreads();

    // ---- issue next tile's global loads + conversion (overlap with WMMAs) ----
    if (k0 + TILE_K < K) load_tile(k0 + TILE_K);

    // ---- A fragment: lane l -> M = l&15, kb = (l>>4)*16;
    //      elems 0..15 = K kb..kb+15
    const int m  = lane & 15;
    const int kb = (lane >> 4) * 16;
    v16bf afrag = cat8(*(const v8bf*)&As[mrow + m][kb],
                       *(const v8bf*)&As[mrow + m][kb + 8]);

    // ---- B fragments (hoisted): lane l -> N = l&15, kb2 = (l>>4)*16
    const int n   = lane & 15;
    const int kb2 = (lane >> 4) * 16;
    v16bf bf0 = cat8(*(const v8bf*)&Bs[n][kb2],      *(const v8bf*)&Bs[n][kb2 + 8]);
    v16bf bf1 = cat8(*(const v8bf*)&Bs[16 + n][kb2], *(const v8bf*)&Bs[16 + n][kb2 + 8]);
    v16bf bf2 = cat8(*(const v8bf*)&Bs[32 + n][kb2], *(const v8bf*)&Bs[32 + n][kb2 + 8]);
    v16bf bf3 = cat8(*(const v8bf*)&Bs[48 + n][kb2], *(const v8bf*)&Bs[48 + n][kb2 + 8]);

    acc0 = __builtin_amdgcn_wmma_f32_16x16x32_bf16(false, afrag, false, bf0, (short)0, acc0, false, false);
    acc1 = __builtin_amdgcn_wmma_f32_16x16x32_bf16(false, afrag, false, bf1, (short)0, acc1, false, false);
    acc2 = __builtin_amdgcn_wmma_f32_16x16x32_bf16(false, afrag, false, bf2, (short)0, acc2, false, false);
    acc3 = __builtin_amdgcn_wmma_f32_16x16x32_bf16(false, afrag, false, bf3, (short)0, acc3, false, false);

    __syncthreads();
  }

  // ---- epilogue: D layout: lane l -> N = l&15, VGPR v -> M = v + (l>>4)*8 ----
  const int nloc  = lane & 15;
  const int mhalf = (lane >> 4) * 8;
  const long long rowbase = (long long)(blockM + mrow + mhalf) * ldc;

#define EPILOGUE(ACC, NC)                                                      \
  {                                                                            \
    const int gn = blockN + (NC) * 16 + nloc;                                  \
    if (!GUARD || gn < N) {                                                    \
      const float bv = bias ? bias[gn] : 0.0f;                                 \
      _Pragma("unroll")                                                        \
      for (int v = 0; v < 8; ++v) {                                            \
        const long long off = rowbase + (long long)v * ldc + gn;               \
        float r = (ACC)[v] * scale + bv;                                       \
        if (relu) r = fmaxf(r, 0.0f);                                          \
        if (Res)  r += Res[off];                                               \
        Cp[off] = r;                                                           \
      }                                                                        \
    }                                                                          \
  }
  EPILOGUE(acc0, 0)
  EPILOGUE(acc1, 1)
  EPILOGUE(acc2, 2)
  EPILOGUE(acc3, 3)
#undef EPILOGUE
}

// ---------------------------------------------------------------------------
// LayerNorm (ddof=1):  out = g * (x - mean) / (sqrt(var) + eps) + b
// ---------------------------------------------------------------------------
__global__ __launch_bounds__(256)
void layernorm_kernel(const float* __restrict__ x, const float* __restrict__ g,
                      const float* __restrict__ b, float* __restrict__ out, int d)
{
  __shared__ float red[256];
  const long long row = blockIdx.x;
  const float* xr = x + row * d;
  float s = 0.f;
  for (int i = threadIdx.x; i < d; i += 256) s += xr[i];
  red[threadIdx.x] = s; __syncthreads();
  for (int off = 128; off > 0; off >>= 1) {
    if ((int)threadIdx.x < off) red[threadIdx.x] += red[threadIdx.x + off];
    __syncthreads();
  }
  const float mean = red[0] / d;
  __syncthreads();
  float sq = 0.f;
  for (int i = threadIdx.x; i < d; i += 256) { float t = xr[i] - mean; sq += t * t; }
  red[threadIdx.x] = sq; __syncthreads();
  for (int off = 128; off > 0; off >>= 1) {
    if ((int)threadIdx.x < off) red[threadIdx.x] += red[threadIdx.x + off];
    __syncthreads();
  }
  const float var = red[0] / (float)(d - 1);
  const float inv = 1.0f / (sqrtf(var) + LN_EPS);
  for (int i = threadIdx.x; i < d; i += 256)
    out[row * d + i] = g[i] * (xr[i] - mean) * inv + b[i];
}

// ---------------------------------------------------------------------------
// Softmax over last dim, in place; optional causal mask (col > row -> -1e9)
// ---------------------------------------------------------------------------
__global__ __launch_bounds__(256)
void softmax_kernel(float* __restrict__ s, int cols, int rowsPerMat, int causal)
{
  __shared__ float red[256];
  const long long row = blockIdx.x;
  const int qi = (int)(row % rowsPerMat);
  float* sr = s + row * (long long)cols;
  float mx = -3.402823e38f;
  for (int i = threadIdx.x; i < cols; i += 256) {
    float v = sr[i];
    if (causal && i > qi) v = -1.0e9f;
    sr[i] = v;
    mx = fmaxf(mx, v);
  }
  red[threadIdx.x] = mx; __syncthreads();
  for (int off = 128; off > 0; off >>= 1) {
    if ((int)threadIdx.x < off) red[threadIdx.x] = fmaxf(red[threadIdx.x], red[threadIdx.x + off]);
    __syncthreads();
  }
  mx = red[0]; __syncthreads();
  float sum = 0.f;
  for (int i = threadIdx.x; i < cols; i += 256) { float e = expf(sr[i] - mx); sr[i] = e; sum += e; }
  red[threadIdx.x] = sum; __syncthreads();
  for (int off = 128; off > 0; off >>= 1) {
    if ((int)threadIdx.x < off) red[threadIdx.x] += red[threadIdx.x + off];
    __syncthreads();
  }
  const float inv = 1.0f / red[0];
  for (int i = threadIdx.x; i < cols; i += 256) sr[i] *= inv;
}

// ---------------------------------------------------------------------------
// y[b,t,:] = emb[tgt[b,t]] * sqrt(D) + PE(t,:)   with trunc(sin)/trunc(cos)
// ---------------------------------------------------------------------------
__global__ void embed_kernel(const int* __restrict__ tgt, const float* __restrict__ emb,
                             float* __restrict__ y, int T, int d)
{
  const long long idx = (long long)blockIdx.x * blockDim.x + threadIdx.x;
  const long long total = (long long)BATCH * T * d;
  if (idx >= total) return;
  const int dd = (int)(idx % d);
  const long long bt = idx / d;
  const int t = (int)(bt % T);
  const int tok = tgt[bt];
  float val = emb[(long long)tok * d + dd] * sqrtf((float)d);
  const int i = dd >> 1;
  const float div = expf((float)(2 * i) * (-logf(10000.0f) / (float)d));
  const float ang = (float)t * div;
  const float pe = (dd & 1) ? truncf(cosf(ang)) : truncf(sinf(ang));
  y[idx] = val + pe;
}

// ---------------------------------------------------------------------------
// In-place log_softmax over last dim
// ---------------------------------------------------------------------------
__global__ __launch_bounds__(256)
void logsoftmax_kernel(float* __restrict__ x, int cols)
{
  __shared__ float red[256];
  const long long row = blockIdx.x;
  float* xr = x + row * (long long)cols;
  float mx = -3.402823e38f;
  for (int i = threadIdx.x; i < cols; i += 256) mx = fmaxf(mx, xr[i]);
  red[threadIdx.x] = mx; __syncthreads();
  for (int off = 128; off > 0; off >>= 1) {
    if ((int)threadIdx.x < off) red[threadIdx.x] = fmaxf(red[threadIdx.x], red[threadIdx.x + off]);
    __syncthreads();
  }
  mx = red[0]; __syncthreads();
  float sum = 0.f;
  for (int i = threadIdx.x; i < cols; i += 256) sum += expf(xr[i] - mx);
  red[threadIdx.x] = sum; __syncthreads();
  for (int off = 128; off > 0; off >>= 1) {
    if ((int)threadIdx.x < off) red[threadIdx.x] += red[threadIdx.x + off];
    __syncthreads();
  }
  const float lse = mx + logf(red[0]);
  for (int i = threadIdx.x; i < cols; i += 256) xr[i] -= lse;
}

// ---------------------------------------------------------------------------
// Host orchestration
// ---------------------------------------------------------------------------
static inline void gemm(hipStream_t st,
                        const float* A, int lda, long long sAb, long long sAh,
                        const float* Bm, int ldb, long long sBb, long long sBh,
                        float* C, int ldc, long long sCb, long long sCh,
                        const float* bias, const float* Res,
                        int M, int N, int K, int bB, int bH,
                        float scale, int transB, int relu)
{
  dim3 g((N + TILE_N - 1) / TILE_N, (M + TILE_M - 1) / TILE_M, bB * bH);
  const bool guard = (N % TILE_N) != 0;
  if (transB) {
    if (guard)
      gemm_wmma_kernel<1, 1><<<g, 256, 0, st>>>(A, lda, sAb, sAh, Bm, ldb, sBb, sBh,
                                                C, ldc, sCb, sCh, bias, Res,
                                                M, N, K, bH, scale, relu);
    else
      gemm_wmma_kernel<1, 0><<<g, 256, 0, st>>>(A, lda, sAb, sAh, Bm, ldb, sBb, sBh,
                                                C, ldc, sCb, sCh, bias, Res,
                                                M, N, K, bH, scale, relu);
  } else {
    gemm_wmma_kernel<0, 0><<<g, 256, 0, st>>>(A, lda, sAb, sAh, Bm, ldb, sBb, sBh,
                                              C, ldc, sCb, sCh, bias, Res,
                                              M, N, K, bH, scale, relu);
  }
}

// Input flattening assumed in setup_inputs() dict-insertion order, recursive:
//  0 att_feats [4,512,2048]   1 tgt [4,512] (int)   2 tgt_masks (unused; causal computed)
//  3 att_w [512,2048]  4 att_b  5 emb [10000,512]  6 gen_w [10000,512]  7 gen_b
//  enc layer l at 8+16*l : wq wk wv wo bq bk bv bo  w1 b1 w2 b2  n1g n1b n2g n2b
//  dec layer l at 104+26*l: sa(wq wk wv wo bq bk bv bo) ca(+8..15 same order)
//                            w1 b1 w2 b2 (+16..19)  n1g n1b n2g n2b n3g n3b (+20..25)
//  260 enc_ng  261 enc_nb  262 dec_ng  263 dec_nb
extern "C" void kernel_launch(void* const* d_in, const int* in_sizes, int n_in,
                              void* d_out, int out_size, void* d_ws, size_t ws_size,
                              hipStream_t stream)
{
  (void)in_sizes; (void)n_in; (void)out_size; (void)ws_size;
  auto F = [&](int i) -> const float* { return (const float*)d_in[i]; };
  const int* tgt = (const int*)d_in[1];

  float* ws = (float*)d_ws;
  const long long NM = (long long)BATCH * SEQ * D_MODEL;   // 1,048,576 floats
  float* X0   = ws;
  float* X1   = X0 + NM;
  float* Hb   = X1 + NM;
  float* Qb   = Hb + NM;
  float* Kb   = Qb + NM;
  float* Vb   = Kb + NM;
  float* Ob   = Vb + NM;
  float* Memb = Ob + NM;
  float* Sc   = Memb + NM;                                      // [B*H, S, S]
  float* Fb   = Sc + (long long)BATCH * N_HEADS * SEQ * SEQ;    // [B*S, FF]

  const int BS = BATCH * SEQ;     // 2048 rows
  const long long sQKV_b = (long long)SEQ * D_MODEL;   // batch stride in [B*S, D] bufs
  const long long sQKV_h = DK_HEAD;                    // head stride (column offset)
  const long long sSc_b  = (long long)N_HEADS * SEQ * SEQ;
  const long long sSc_h  = (long long)SEQ * SEQ;

  // ---- input projection: x = relu(att_feats @ att_w^T + att_b) ----
  gemm(stream, F(0), ATT_DIM, 0, 0, F(3), ATT_DIM, 0, 0, X0, D_MODEL, 0, 0,
       F(4), nullptr, BS, D_MODEL, ATT_DIM, 1, 1, 1.0f, 1, 1);

  float* x = X0; float* xa = X1;

  // ---- encoder ----
  for (int l = 0; l < NLAYER; ++l) {
    const int p = 8 + 16 * l;
    // self-attention
    layernorm_kernel<<<BS, 256, 0, stream>>>(x, F(p + 12), F(p + 13), Hb, D_MODEL);
    gemm(stream, Hb, D_MODEL, 0, 0, F(p + 0), D_MODEL, 0, 0, Qb, D_MODEL, 0, 0,
         F(p + 4), nullptr, BS, D_MODEL, D_MODEL, 1, 1, 1.0f, 1, 0);
    gemm(stream, Hb, D_MODEL, 0, 0, F(p + 1), D_MODEL, 0, 0, Kb, D_MODEL, 0, 0,
         F(p + 5), nullptr, BS, D_MODEL, D_MODEL, 1, 1, 1.0f, 1, 0);
    gemm(stream, Hb, D_MODEL, 0, 0, F(p + 2), D_MODEL, 0, 0, Vb, D_MODEL, 0, 0,
         F(p + 6), nullptr, BS, D_MODEL, D_MODEL, 1, 1, 1.0f, 1, 0);
    gemm(stream, Qb, D_MODEL, sQKV_b, sQKV_h, Kb, D_MODEL, sQKV_b, sQKV_h,
         Sc, SEQ, sSc_b, sSc_h, nullptr, nullptr,
         SEQ, SEQ, DK_HEAD, BATCH, N_HEADS, 0.125f, 1, 0);
    softmax_kernel<<<BATCH * N_HEADS * SEQ, 256, 0, stream>>>(Sc, SEQ, SEQ, 0);
    gemm(stream, Sc, SEQ, sSc_b, sSc_h, Vb, D_MODEL, sQKV_b, sQKV_h,
         Ob, D_MODEL, sQKV_b, sQKV_h, nullptr, nullptr,
         SEQ, DK_HEAD, SEQ, BATCH, N_HEADS, 1.0f, 0, 0);
    gemm(stream, Ob, D_MODEL, 0, 0, F(p + 3), D_MODEL, 0, 0, xa, D_MODEL, 0, 0,
         F(p + 7), x, BS, D_MODEL, D_MODEL, 1, 1, 1.0f, 1, 0);
    { float* t2 = x; x = xa; xa = t2; }
    // FFN
    layernorm_kernel<<<BS, 256, 0, stream>>>(x, F(p + 14), F(p + 15), Hb, D_MODEL);
    gemm(stream, Hb, D_MODEL, 0, 0, F(p + 8), D_MODEL, 0, 0, Fb, FF_DIM, 0, 0,
         F(p + 9), nullptr, BS, FF_DIM, D_MODEL, 1, 1, 1.0f, 1, 1);
    gemm(stream, Fb, FF_DIM, 0, 0, F(p + 10), FF_DIM, 0, 0, xa, D_MODEL, 0, 0,
         F(p + 11), x, BS, D_MODEL, FF_DIM, 1, 1, 1.0f, 1, 0);
    { float* t2 = x; x = xa; xa = t2; }
  }
  layernorm_kernel<<<BS, 256, 0, stream>>>(x, F(260), F(261), Memb, D_MODEL);

  // ---- decoder ----
  float* y = X0; float* ya = X1;
  embed_kernel<<<(int)((NM + 255) / 256), 256, 0, stream>>>(tgt, F(5), y, SEQ, D_MODEL);

  for (int l = 0; l < NLAYER; ++l) {
    const int p = 104 + 26 * l;
    // masked self-attention
    layernorm_kernel<<<BS, 256, 0, stream>>>(y, F(p + 20), F(p + 21), Hb, D_MODEL);
    gemm(stream, Hb, D_MODEL, 0, 0, F(p + 0), D_MODEL, 0, 0, Qb, D_MODEL, 0, 0,
         F(p + 4), nullptr, BS, D_MODEL, D_MODEL, 1, 1, 1.0f, 1, 0);
    gemm(stream, Hb, D_MODEL, 0, 0, F(p + 1), D_MODEL, 0, 0, Kb, D_MODEL, 0, 0,
         F(p + 5), nullptr, BS, D_MODEL, D_MODEL, 1, 1, 1.0f, 1, 0);
    gemm(stream, Hb, D_MODEL, 0, 0, F(p + 2), D_MODEL, 0, 0, Vb, D_MODEL, 0, 0,
         F(p + 6), nullptr, BS, D_MODEL, D_MODEL, 1, 1, 1.0f, 1, 0);
    gemm(stream, Qb, D_MODEL, sQKV_b, sQKV_h, Kb, D_MODEL, sQKV_b, sQKV_h,
         Sc, SEQ, sSc_b, sSc_h, nullptr, nullptr,
         SEQ, SEQ, DK_HEAD, BATCH, N_HEADS, 0.125f, 1, 0);
    softmax_kernel<<<BATCH * N_HEADS * SEQ, 256, 0, stream>>>(Sc, SEQ, SEQ, 1);
    gemm(stream, Sc, SEQ, sSc_b, sSc_h, Vb, D_MODEL, sQKV_b, sQKV_h,
         Ob, D_MODEL, sQKV_b, sQKV_h, nullptr, nullptr,
         SEQ, DK_HEAD, SEQ, BATCH, N_HEADS, 1.0f, 0, 0);
    gemm(stream, Ob, D_MODEL, 0, 0, F(p + 3), D_MODEL, 0, 0, ya, D_MODEL, 0, 0,
         F(p + 7), y, BS, D_MODEL, D_MODEL, 1, 1, 1.0f, 1, 0);
    { float* t2 = y; y = ya; ya = t2; }
    // cross-attention (K,V from encoder memory)
    layernorm_kernel<<<BS, 256, 0, stream>>>(y, F(p + 22), F(p + 23), Hb, D_MODEL);
    gemm(stream, Hb, D_MODEL, 0, 0, F(p + 8), D_MODEL, 0, 0, Qb, D_MODEL, 0, 0,
         F(p + 12), nullptr, BS, D_MODEL, D_MODEL, 1, 1, 1.0f, 1, 0);
    gemm(stream, Memb, D_MODEL, 0, 0, F(p + 9), D_MODEL, 0, 0, Kb, D_MODEL, 0, 0,
         F(p + 13), nullptr, BS, D_MODEL, D_MODEL, 1, 1, 1.0f, 1, 0);
    gemm(stream, Memb, D_MODEL, 0, 0, F(p + 10), D_MODEL, 0, 0, Vb, D_MODEL, 0, 0,
         F(p + 14), nullptr, BS, D_MODEL, D_MODEL, 1, 1, 1.0f, 1, 0);
    gemm(stream, Qb, D_MODEL, sQKV_b, sQKV_h, Kb, D_MODEL, sQKV_b, sQKV_h,
         Sc, SEQ, sSc_b, sSc_h, nullptr, nullptr,
         SEQ, SEQ, DK_HEAD, BATCH, N_HEADS, 0.125f, 1, 0);
    softmax_kernel<<<BATCH * N_HEADS * SEQ, 256, 0, stream>>>(Sc, SEQ, SEQ, 0);
    gemm(stream, Sc, SEQ, sSc_b, sSc_h, Vb, D_MODEL, sQKV_b, sQKV_h,
         Ob, D_MODEL, sQKV_b, sQKV_h, nullptr, nullptr,
         SEQ, DK_HEAD, SEQ, BATCH, N_HEADS, 1.0f, 0, 0);
    gemm(stream, Ob, D_MODEL, 0, 0, F(p + 11), D_MODEL, 0, 0, ya, D_MODEL, 0, 0,
         F(p + 15), y, BS, D_MODEL, D_MODEL, 1, 1, 1.0f, 1, 0);
    { float* t2 = y; y = ya; ya = t2; }
    // FFN
    layernorm_kernel<<<BS, 256, 0, stream>>>(y, F(p + 24), F(p + 25), Hb, D_MODEL);
    gemm(stream, Hb, D_MODEL, 0, 0, F(p + 16), D_MODEL, 0, 0, Fb, FF_DIM, 0, 0,
         F(p + 17), nullptr, BS, FF_DIM, D_MODEL, 1, 1, 1.0f, 1, 1);
    gemm(stream, Fb, FF_DIM, 0, 0, F(p + 18), FF_DIM, 0, 0, ya, D_MODEL, 0, 0,
         F(p + 19), y, BS, D_MODEL, FF_DIM, 1, 1, 1.0f, 1, 0);
    { float* t2 = y; y = ya; ya = t2; }
  }

  // ---- generator + log_softmax ----
  layernorm_kernel<<<BS, 256, 0, stream>>>(y, F(262), F(263), Hb, D_MODEL);
  gemm(stream, Hb, D_MODEL, 0, 0, F(6), D_MODEL, 0, 0, (float*)d_out, VOCAB, 0, 0,
       F(7), nullptr, BS, VOCAB, D_MODEL, 1, 1, 1.0f, 1, 0);
  logsoftmax_kernel<<<BS, 256, 0, stream>>>((float*)d_out, VOCAB);
}
